// RNN_Classifier_17128329576832
// MI455X (gfx1250) — matrix operand order repaired and checked by reading, not compiled, not run on hardware
//
#include <hip/hip_runtime.h>
#include <hip/hip_bf16.h>

#define BATCH 32768
#define TT    28
#define INF   28
#define HH    50
#define NL    5
#define NCLS  10
#define HP    64
#define TILE_B 16

typedef __bf16 bf16;
typedef bf16  v16bf __attribute__((ext_vector_type(16)));
typedef float v8f   __attribute__((ext_vector_type(8)));

// RNE pack: used only for the one-time weight conversion.
static __device__ __forceinline__ bf16 f2bf_rne(float f) {
  union { float f; unsigned u; } v; v.f = f;
  unsigned u = v.u + 0x7FFFu + ((v.u >> 16) & 1u);
  unsigned short h = (unsigned short)(u >> 16);
  bf16 b; __builtin_memcpy(&b, &h, 2); return b;
}
// Truncating pack: hot path (single shift), fine for tanh-squashed activations.
static __device__ __forceinline__ unsigned short f2bf_trunc_bits(float f) {
  union { float f; unsigned u; } v; v.f = f;
  return (unsigned short)(v.u >> 16);
}
static __device__ __forceinline__ bf16 f2bf_trunc(float f) {
  unsigned short h = f2bf_trunc_bits(f);
  bf16 b; __builtin_memcpy(&b, &h, 2); return b;
}
static __device__ __forceinline__ float bf2f(unsigned short h) {
  union { unsigned u; float f; } v; v.u = ((unsigned)h) << 16; return v.f;
}

// Branchless tanh on the hardware transcendental unit; tanh_fast(0) == 0 exactly,
// so zero-padded columns stay exactly zero with no predicate needed.
static __device__ __forceinline__ float tanh_fast(float x) {
  float ax = __builtin_fabsf(x);
  float e  = __builtin_amdgcn_exp2f(ax * -2.885390081777927f);
  float t  = (1.0f - e) * __builtin_amdgcn_rcpf(1.0f + e);
  return __builtin_copysignf(t, x);
}

// A-operand (16-bit 16x32 layout): lane<16 holds K {kb0..+7, kb1..+7}, lane>=16 the
// complementary chunks. Two ds_load_b128 per 32-K step.
static __device__ __forceinline__ void loadA(const unsigned short* row, int kb0, int kb1,
                                             v16bf& lo, v16bf& hi) {
  uint4 q0 = *(const uint4*)(row + kb0);
  uint4 q1 = *(const uint4*)(row + kb1);
  uint4 q2 = *(const uint4*)(row + 32 + kb0);
  uint4 q3 = *(const uint4*)(row + 32 + kb1);
  __builtin_memcpy(&lo, &q0, 16); __builtin_memcpy((char*)&lo + 16, &q1, 16);
  __builtin_memcpy(&hi, &q2, 16); __builtin_memcpy((char*)&hi + 16, &q3, 16);
}

__global__ __launch_bounds__(128, 1)
void rnn5_wmma_fused(const float* __restrict__ x,   const float* __restrict__ Wih0,
                     const float* __restrict__ Wih, const float* __restrict__ Whh,
                     const float* __restrict__ bih, const float* __restrict__ bhh,
                     const float* __restrict__ Wfc, const float* __restrict__ bfc,
                     float* __restrict__ out)
{
  // Whole time series for this 16-row batch tile (bf16, H padded to 64): 56 KB.
  // act[:,t,:] holds the previous layer's output until step t overwrites it with
  // h_t, so act[:,t-1,:] is exactly h_{t-1} for the recurrent term.
  __shared__ __align__(16) unsigned short act[TILE_B][TT][HP];

  const int tid   = threadIdx.x;
  const int lane  = tid & 31;
  const int ntile = tid >> 5;            // wave id -> hidden-column tile
  const int half  = (lane >> 4) & 1;
  const int l15   = lane & 15;
  const int b0    = blockIdx.x * TILE_B;

  const int aM   = l15;                  // A row (local batch index)
  const int akb0 = half ? 16 : 0;        // A K-chunk bases
  const int akb1 = half ? 24 : 8;
  const int bkb  = half ? 16 : 0;        // B K base
  const int ng   = ntile * 16 + l15;     // global hidden column owned in B/D
  const int mrow = half ? 8 : 0;         // D-tile row base for this lane
  const bool ngv = (ng < HH);

  // =========================== Layer 0 (input from x) ===========================
  {
    v16bf Bih0, Bhh[2];
    #pragma unroll
    for (int e = 0; e < 16; ++e) {
      int k = bkb + e;
      Bih0[e] = f2bf_rne((ngv && k < INF) ? Wih0[ng * INF + k] : 0.0f);  // B[k,n]=W[n,k]
    }
    #pragma unroll
    for (int ks = 0; ks < 2; ++ks)
      #pragma unroll
      for (int e = 0; e < 16; ++e) {
        int k = ks * 32 + bkb + e;
        Bhh[ks][e] = f2bf_rne((ngv && k < HH) ? Whh[ng * HH + k] : 0.0f);
      }
    const float bias = ngv ? (bih[ng] + bhh[ng]) : 0.0f;  // pad columns: bias==0

    for (int t = 0; t < TT; ++t) {
      // ---- A_ih straight from HBM (read exactly once), f32 -> bf16, K pad 32 ----
      const float* xr = x + ((size_t)(b0 + aM) * TT + t) * INF;
      if (t + 1 < TT) __builtin_prefetch(xr + INF, 0, 0);
      v16bf Aih = {};
      float4 q0 = *(const float4*)(xr + akb0);       // k akb0..+3  (<= 19)
      float4 q1 = *(const float4*)(xr + akb0 + 4);   // k akb0+4..+7 (<= 23)
      float4 q2 = *(const float4*)(xr + akb1);       // k akb1..+3  (<= 27)
      Aih[0] = f2bf_trunc(q0.x); Aih[1] = f2bf_trunc(q0.y);
      Aih[2] = f2bf_trunc(q0.z); Aih[3] = f2bf_trunc(q0.w);
      Aih[4] = f2bf_trunc(q1.x); Aih[5] = f2bf_trunc(q1.y);
      Aih[6] = f2bf_trunc(q1.z); Aih[7] = f2bf_trunc(q1.w);
      Aih[8]  = f2bf_trunc(q2.x); Aih[9]  = f2bf_trunc(q2.y);
      Aih[10] = f2bf_trunc(q2.z); Aih[11] = f2bf_trunc(q2.w);
      #pragma unroll
      for (int e = 12; e < 16; ++e) {                // k akb1+4..+7 (may exceed 27)
        int k = akb1 + e - 8;
        float v = xr[k < INF ? k : 0];               // clamped: never reads OOB
        Aih[e] = f2bf_trunc(k < INF ? v : 0.0f);
      }

      v8f c;
      #pragma unroll
      for (int r = 0; r < 8; ++r) c[r] = bias;
      c = __builtin_amdgcn_wmma_f32_16x16x32_bf16(false, Aih, false, Bih0, (short)0, c, false, false);

      if (t > 0) {   // h_{-1}==0; reads act[t-1], writes act[t] -> no pre-barrier needed
        v16bf Ahh0, Ahh1;
        loadA(&act[aM][t - 1][0], akb0, akb1, Ahh0, Ahh1);
        c = __builtin_amdgcn_wmma_f32_16x16x32_bf16(false, Ahh0, false, Bhh[0], (short)0, c, false, false);
        c = __builtin_amdgcn_wmma_f32_16x16x32_bf16(false, Ahh1, false, Bhh[1], (short)0, c, false, false);
      }

      // Pad columns have all-zero B and bias -> c==0 -> tanh_fast(0)==0:
      // store unconditionally, no predicates, no exec masking.
      #pragma unroll
      for (int r = 0; r < 8; ++r)
        act[mrow + r][t][ng] = f2bf_trunc_bits(tanh_fast(c[r]));
      __syncthreads();                               // h_t visible before step t+1
    }
  }

  // =========================== Layers 1..4 (input in LDS) =======================
  for (int l = 1; l < NL; ++l) {
    const float* wih = Wih + (size_t)(l - 1) * HH * HH;
    const float* whh = Whh + (size_t)l * HH * HH;

    v16bf Bih[2], Bhh[2];
    #pragma unroll
    for (int ks = 0; ks < 2; ++ks)
      #pragma unroll
      for (int e = 0; e < 16; ++e) {
        int k = ks * 32 + bkb + e;
        bool kv = (k < HH) & ngv;
        Bih[ks][e] = f2bf_rne(kv ? wih[ng * HH + k] : 0.0f);
        Bhh[ks][e] = f2bf_rne(kv ? whh[ng * HH + k] : 0.0f);
      }
    const float bias = ngv ? (bih[l * HH + ng] + bhh[l * HH + ng]) : 0.0f;

    // ---- t = 0: no recurrent term ----
    {
      v16bf Aih0, Aih1;
      loadA(&act[aM][0][0], akb0, akb1, Aih0, Aih1);
      __syncthreads();                               // reads of act[0] done before overwrite
      v8f c;
      #pragma unroll
      for (int r = 0; r < 8; ++r) c[r] = bias;
      c = __builtin_amdgcn_wmma_f32_16x16x32_bf16(false, Aih0, false, Bih[0], (short)0, c, false, false);
      c = __builtin_amdgcn_wmma_f32_16x16x32_bf16(false, Aih1, false, Bih[1], (short)0, c, false, false);
      #pragma unroll
      for (int r = 0; r < 8; ++r)
        act[mrow + r][0][ng] = f2bf_trunc_bits(tanh_fast(c[r]));
      __syncthreads();
    }

    // ---- steady state: straight-line body, 4 WMMAs, 2 barriers ----
    for (int t = 1; t < TT; ++t) {
      v16bf Aih0, Aih1, Ahh0, Ahh1;
      loadA(&act[aM][t][0],     akb0, akb1, Aih0, Aih1);
      loadA(&act[aM][t - 1][0], akb0, akb1, Ahh0, Ahh1);
      __syncthreads();                               // reads done before act[t] overwrite

      v8f c;
      #pragma unroll
      for (int r = 0; r < 8; ++r) c[r] = bias;
      c = __builtin_amdgcn_wmma_f32_16x16x32_bf16(false, Aih0, false, Bih[0], (short)0, c, false, false);
      c = __builtin_amdgcn_wmma_f32_16x16x32_bf16(false, Aih1, false, Bih[1], (short)0, c, false, false);
      c = __builtin_amdgcn_wmma_f32_16x16x32_bf16(false, Ahh0, false, Bhh[0], (short)0, c, false, false);
      c = __builtin_amdgcn_wmma_f32_16x16x32_bf16(false, Ahh1, false, Bhh[1], (short)0, c, false, false);

      #pragma unroll
      for (int r = 0; r < 8; ++r)
        act[mrow + r][t][ng] = f2bf_trunc_bits(tanh_fast(c[r]));
      __syncthreads();                               // h_t visible before step t+1
    }
  }

  // ---- final classifier on h_{T-1} of the top layer (160 dots of length 50) ----
  for (int idx = tid; idx < TILE_B * NCLS; idx += (int)blockDim.x) {
    int bl = idx / NCLS, cc = idx % NCLS;
    float s = bfc[cc];
    #pragma unroll 5
    for (int h = 0; h < HH; ++h)
      s += bf2f(act[bl][TT - 1][h]) * Wfc[cc * HH + h];
    out[(size_t)(b0 + bl) * NCLS + cc] = s;
  }
}

extern "C" void kernel_launch(void* const* d_in, const int* in_sizes, int n_in,
                              void* d_out, int out_size, void* d_ws, size_t ws_size,
                              hipStream_t stream) {
  const float* x    = (const float*)d_in[0];
  const float* Wih0 = (const float*)d_in[1];
  const float* Wih  = (const float*)d_in[2];
  const float* Whh  = (const float*)d_in[3];
  const float* bih  = (const float*)d_in[4];
  const float* bhh  = (const float*)d_in[5];
  const float* Wfc  = (const float*)d_in[6];
  const float* bfc  = (const float*)d_in[7];
  float* out = (float*)d_out;

  dim3 grid(BATCH / TILE_B);   // 2048 workgroups, 16 batch rows each
  dim3 block(128);             // 4 wave32s, one hidden-column tile each
  hipLaunchKernelGGL(rnn5_wmma_fused, grid, block, 0, stream,
                     x, Wih0, Wih, Whh, bih, bhh, Wfc, bfc, out);
}
